// ImplicitSAGPT_38465727103895
// MI455X (gfx1250) — matrix operand order repaired and checked by reading, not compiled, not run on hardware
//
#include <hip/hip_runtime.h>
#include <hip/hip_bf16.h>

// ---------------------------------------------------------------------------
// ImplicitSAGPT forward for gfx1250 (MI455X).
// All GEMM contractions + attention run through v_wmma_f32_16x16x32_bf16
// (bf16 inputs, f32 accumulation). Flash-style attention avoids an SxS
// score buffer. Weights (~75MB) fit in the 192MB L2.
// ---------------------------------------------------------------------------

typedef __attribute__((ext_vector_type(16))) __bf16 v16bf;
typedef __attribute__((ext_vector_type(8)))  float  v8f;

#define E_DIM  512
#define SEQ    768          // 3*T
#define TT     256
#define BBATCH 8
#define NHEAD  8
#define HDIM   64
#define NLAYER 6

__device__ __forceinline__ v8f wmma_bf16(v16bf a, v16bf b, v8f c) {
  // D = A(16x32 bf16) * B(32x16 bf16) + C(16x16 f32)
  return __builtin_amdgcn_wmma_f32_16x16x32_bf16(false, a, false, b,
                                                 (short)0, c, false, false);
}

__device__ __forceinline__ float act_apply(float v, int act) {
  if (act == 1)      v = fmaxf(v, 0.0f);
  else if (act == 2) v = 0.5f * v * (1.0f + erff(v * 0.70710678118654752f));
  return v;
}

// ---------------------------------------------------------------------------
// WMMA GEMM: C[M,N] = act(A[M,K] @ W[K,N] + bias [+ R])
// EDGE=false fast path: requires K%32==0, N%64==0, M%16==0.
//   One wave computes a 16x64 tile (4 WMMAs share one A operand); no bounds
//   checks anywhere in the K loop -> branch-free, clause-able loads.
// EDGE=true: 16x16 tile per wave, branchless clamped loads (cndmask, no
//   EXEC branches). Used only by tiny encoder/head GEMMs.
// ---------------------------------------------------------------------------
template <bool EDGE>
__global__ __launch_bounds__(256) void gemm_wmma_kernel(
    const float* __restrict__ A, int lda,
    const float* __restrict__ W, const float* __restrict__ bias,
    const float* __restrict__ R,
    float* __restrict__ C, int ldc,
    int M, int N, int K, int act)
{
  const int lane = threadIdx.x & 31;
  const int wave = threadIdx.x >> 5;
  const int half = lane >> 4, l15 = lane & 15;

  if (!EDGE) {
    const int Nt4 = N >> 6;                    // 64-wide column tiles
    const int tile = blockIdx.x * 8 + wave;
    if (tile >= (M >> 4) * Nt4) return;        // wave-uniform
    const int tm = tile / Nt4, tn = tile % Nt4;
    const int m = tm * 16 + l15;
    const int ncol = tn * 64 + l15;            // first of 4 column groups

    v8f acc0 = {0.f,0.f,0.f,0.f,0.f,0.f,0.f,0.f};
    v8f acc1 = acc0, acc2 = acc0, acc3 = acc0;
    const float* Arow = A + (size_t)m * lda;

    for (int kk = 0; kk < K; kk += 32) {
      // A 16x32 bf16 operand (contiguous k = 16*half + element): 4 unguarded float4 loads
      v16bf a;
      #pragma unroll
      for (int g = 0; g < 2; ++g) {
        #pragma unroll
        for (int q4 = 0; q4 < 2; ++q4) {
          const float4 f = *(const float4*)(Arow + kk + (g << 3) + (half << 4) + (q4 << 2));
          const int e0 = (g << 3) + (q4 << 2);
          a[e0+0] = (__bf16)f.x; a[e0+1] = (__bf16)f.y;
          a[e0+2] = (__bf16)f.z; a[e0+3] = (__bf16)f.w;
        }
      }
      // Four B 32x16 operands (columns ncol, +16, +32, +48)
      v16bf b0, b1, b2, b3;
      #pragma unroll
      for (int e = 0; e < 16; ++e) {
        const float* wr = W + (size_t)(kk + (half << 4) + e) * N + ncol;
        b0[e] = (__bf16)wr[0];
        b1[e] = (__bf16)wr[16];
        b2[e] = (__bf16)wr[32];
        b3[e] = (__bf16)wr[48];
      }
      acc0 = wmma_bf16(a, b0, acc0);
      acc1 = wmma_bf16(a, b1, acc1);
      acc2 = wmma_bf16(a, b2, acc2);
      acc3 = wmma_bf16(a, b3, acc3);
    }
    // epilogue
    #pragma unroll
    for (int r = 0; r < 8; ++r) {
      const int mm = tm * 16 + r + (half << 3);
      float* cr = C + (size_t)mm * ldc + ncol;
      const float* rr = R ? (R + (size_t)mm * ldc + ncol) : nullptr;
      float v0 = acc0[r] + bias[ncol];
      float v1 = acc1[r] + bias[ncol + 16];
      float v2 = acc2[r] + bias[ncol + 32];
      float v3 = acc3[r] + bias[ncol + 48];
      if (rr) { v0 += rr[0]; v1 += rr[16]; v2 += rr[32]; v3 += rr[48]; }
      cr[0]  = act_apply(v0, act);
      cr[16] = act_apply(v1, act);
      cr[32] = act_apply(v2, act);
      cr[48] = act_apply(v3, act);
    }
  } else {
    const int Nt = (N + 15) >> 4;
    const int Mt = (M + 15) >> 4;
    const int tile = blockIdx.x * 8 + wave;
    if (tile >= Mt * Nt) return;
    const int tm = tile / Nt, tn = tile % Nt;
    const int m  = tm * 16 + l15;
    const int nc = tn * 16 + l15;
    const int mc = min(m, M - 1);
    const int ncc = min(nc, N - 1);

    v8f acc = {0.f,0.f,0.f,0.f,0.f,0.f,0.f,0.f};
    for (int kk = 0; kk < K; kk += 32) {
      v16bf a, bm;
      #pragma unroll
      for (int g = 0; g < 2; ++g) {
        #pragma unroll
        for (int q4 = 0; q4 < 2; ++q4) {
          const int kb = kk + (g << 3) + (half << 4) + (q4 << 2);
          const bool ok = (m < M) && (kb + 4 <= K);
          // clamped, always-legal load; zero via cndmask (K%4==0 guaranteed)
          float4 f = *(const float4*)(A + (size_t)mc * lda + min(kb, K - 4));
          f.x = ok ? f.x : 0.f; f.y = ok ? f.y : 0.f;
          f.z = ok ? f.z : 0.f; f.w = ok ? f.w : 0.f;
          const int e0 = (g << 3) + (q4 << 2);
          a[e0+0] = (__bf16)f.x; a[e0+1] = (__bf16)f.y;
          a[e0+2] = (__bf16)f.z; a[e0+3] = (__bf16)f.w;
        }
      }
      #pragma unroll
      for (int e = 0; e < 16; ++e) {
        const int kb = kk + (half << 4) + e;
        float bv = W[(size_t)min(kb, K - 1) * N + ncc];
        bv = (kb < K && nc < N) ? bv : 0.0f;
        bm[e] = (__bf16)bv;
      }
      acc = wmma_bf16(a, bm, acc);
    }
    #pragma unroll
    for (int r = 0; r < 8; ++r) {
      const int mm = tm * 16 + r + (half << 3);
      if (mm < M && nc < N) {
        float v = acc[r] + bias[nc];
        if (R) v += R[(size_t)mm * ldc + nc];
        C[(size_t)mm * ldc + nc] = act_apply(v, act);
      }
    }
  }
}

// ---------------------------------------------------------------------------
// Fused causal flash attention: one wave per (b, h, 16-query tile).
// scores via 2x WMMA (HD=64 split into two K=32 chunks); online softmax with
// shfl_xor row reductions; P re-laid out through LDS (s_wait_dscnt) into
// A-operand format for the P@V WMMAs.
// ---------------------------------------------------------------------------
__global__ __launch_bounds__(32) void attn_flash_kernel(
    const float* __restrict__ Q, const float* __restrict__ K,
    const float* __restrict__ V, float* __restrict__ Y)
{
  __shared__ __bf16 Pl[16][32];
  const int lane = threadIdx.x;
  const int qt = blockIdx.x, h = blockIdx.y, b = blockIdx.z;
  const int qbase = qt << 4;
  const int half = lane >> 4, l15 = lane & 15;
  const float scale = 0.125f;   // 1/sqrt(64)

  const float* Qb = Q + (size_t)b * SEQ * E_DIM + h * HDIM;
  const float* Kb = K + (size_t)b * SEQ * E_DIM + h * HDIM;
  const float* Vb = V + (size_t)b * SEQ * E_DIM + h * HDIM;

  // Q tile as two A-operands (hd 0..31 and 32..63), scale folded in
  v16bf aq0, aq1;
  {
    const float* qp = Qb + (size_t)(qbase + l15) * E_DIM;
    #pragma unroll
    for (int g = 0; g < 2; ++g) {
      #pragma unroll
      for (int q4 = 0; q4 < 2; ++q4) {
        const int kofs = (g << 3) + (half << 4) + (q4 << 2);
        float4 f0 = *(const float4*)(qp + kofs);
        float4 f1 = *(const float4*)(qp + 32 + kofs);
        const int e0 = (g << 3) + (q4 << 2);
        aq0[e0+0] = (__bf16)(f0.x*scale); aq0[e0+1] = (__bf16)(f0.y*scale);
        aq0[e0+2] = (__bf16)(f0.z*scale); aq0[e0+3] = (__bf16)(f0.w*scale);
        aq1[e0+0] = (__bf16)(f1.x*scale); aq1[e0+1] = (__bf16)(f1.y*scale);
        aq1[e0+2] = (__bf16)(f1.z*scale); aq1[e0+3] = (__bf16)(f1.w*scale);
      }
    }
  }

  v8f oacc[4];
  {
    v8f z = {0.f,0.f,0.f,0.f,0.f,0.f,0.f,0.f};
    oacc[0] = z; oacc[1] = z; oacc[2] = z; oacc[3] = z;
  }
  float mrow[8], lrow[8];
  #pragma unroll
  for (int r = 0; r < 8; ++r) { mrow[r] = -1e30f; lrow[r] = 0.0f; }

  const int nchunks = (qbase + 47) >> 5;   // covers keys 0..qbase+15
  for (int jc = 0; jc < nchunks; ++jc) {
    const int jbase = jc << 5;
    // two 16x16 score tiles (keys jbase..+15, jbase+16..+31)
    v8f s0, s1;
    #pragma unroll
    for (int t = 0; t < 2; ++t) {
      v16bf bk0, bk1;
      const int key = jbase + (t << 4) + l15;
      const float* kp = Kb + (size_t)key * E_DIM;
      #pragma unroll
      for (int q4 = 0; q4 < 4; ++q4) {
        const int kofs = (half << 4) + (q4 << 2);
        float4 f0 = *(const float4*)(kp + kofs);
        float4 f1 = *(const float4*)(kp + 32 + kofs);
        const int e0 = q4 << 2;
        bk0[e0+0]=(__bf16)f0.x; bk0[e0+1]=(__bf16)f0.y;
        bk0[e0+2]=(__bf16)f0.z; bk0[e0+3]=(__bf16)f0.w;
        bk1[e0+0]=(__bf16)f1.x; bk1[e0+1]=(__bf16)f1.y;
        bk1[e0+2]=(__bf16)f1.z; bk1[e0+3]=(__bf16)f1.w;
      }
      v8f z = {0.f,0.f,0.f,0.f,0.f,0.f,0.f,0.f};
      z = wmma_bf16(aq0, bk0, z);
      z = wmma_bf16(aq1, bk1, z);
      if (t == 0) s0 = z; else s1 = z;
    }
    // causal mask + online softmax, write P to LDS in (row, col) layout
    #pragma unroll
    for (int r = 0; r < 8; ++r) {
      const int qg = qbase + r + (half << 3);
      float v0 = (jbase + l15      <= qg) ? s0[r] : -1e30f;
      float v1 = (jbase + 16 + l15 <= qg) ? s1[r] : -1e30f;
      float rm = fmaxf(v0, v1);
      #pragma unroll
      for (int off = 1; off < 16; off <<= 1)
        rm = fmaxf(rm, __shfl_xor(rm, off, 32));
      const float newm  = fmaxf(mrow[r], rm);
      const float alpha = __expf(mrow[r] - newm);
      const float p0 = __expf(v0 - newm);
      const float p1 = __expf(v1 - newm);
      float ps = p0 + p1;
      #pragma unroll
      for (int off = 1; off < 16; off <<= 1)
        ps += __shfl_xor(ps, off, 32);
      lrow[r] = lrow[r] * alpha + ps;
      mrow[r] = newm;
      oacc[0][r] *= alpha; oacc[1][r] *= alpha;
      oacc[2][r] *= alpha; oacc[3][r] *= alpha;
      Pl[r + (half << 3)][l15]      = (__bf16)p0;
      Pl[r + (half << 3)][16 + l15] = (__bf16)p1;
    }
    asm volatile("s_wait_dscnt 0x0" ::: "memory");   // cross-lane LDS handoff
    // P (16x32) as A-operand from LDS
    v16bf ap;
    #pragma unroll
    for (int e = 0; e < 16; ++e) {
      const int k = (half << 4) + e;
      ap[e] = Pl[l15][k];
    }
    // O += P @ V  (V chunk 32x64 as four 32x16 B-operands)
    #pragma unroll
    for (int nt = 0; nt < 4; ++nt) {
      v16bf bv;
      #pragma unroll
      for (int e = 0; e < 16; ++e) {
        const int kv = jbase + (half << 4) + e;
        bv[e] = (__bf16)Vb[(size_t)kv * E_DIM + (nt << 4) + l15];
      }
      oacc[nt] = wmma_bf16(ap, bv, oacc[nt]);
    }
  }
  // normalize and write Y[b, q, h, hd]
  #pragma unroll
  for (int r = 0; r < 8; ++r) {
    const float inv = 1.0f / lrow[r];
    const int row = qbase + r + (half << 3);
    float* yp = Y + ((size_t)b * SEQ + row) * E_DIM + h * HDIM;
    yp[l15]      = oacc[0][r] * inv;
    yp[16 + l15] = oacc[1][r] * inv;
    yp[32 + l15] = oacc[2][r] * inv;
    yp[48 + l15] = oacc[3][r] * inv;
  }
}

// ---------------------------------------------------------------------------
// LayerNorm over E=512 (one 128-thread block per row; works in-place)
// ---------------------------------------------------------------------------
__global__ __launch_bounds__(128) void ln_kernel(
    const float* __restrict__ X, int ldx,
    const float* __restrict__ w, const float* __restrict__ b,
    float* __restrict__ O, int ldo)
{
  __shared__ float ss[128], sq[128];
  const int row = blockIdx.x, tid = threadIdx.x;
  const float* xr = X + (size_t)row * ldx;
  float v[4]; float s = 0.f, q = 0.f;
  #pragma unroll
  for (int i = 0; i < 4; ++i) { v[i] = xr[tid + 128*i]; s += v[i]; q += v[i]*v[i]; }
  ss[tid] = s; sq[tid] = q; __syncthreads();
  for (int st = 64; st > 0; st >>= 1) {
    if (tid < st) { ss[tid] += ss[tid + st]; sq[tid] += sq[tid + st]; }
    __syncthreads();
  }
  const float mean = ss[0] * (1.0f / 512.0f);
  const float var  = sq[0] * (1.0f / 512.0f) - mean * mean;
  const float inv  = rsqrtf(var + 1e-5f);
  #pragma unroll
  for (int i = 0; i < 4; ++i) {
    const int c = tid + 128*i;
    O[(size_t)row * ldo + c] = (v[i] - mean) * inv * w[c] + b[c];
  }
}

// x = tok + gpe[timestep] + lpe[t]; zero the t=T-1 action token first
__global__ void embed_kernel(float* __restrict__ X,
                             const float* __restrict__ gpe,
                             const float* __restrict__ lpe,
                             const int* __restrict__ ts)
{
  const int idx = blockIdx.x * 256 + threadIdx.x;
  if (idx >= BBATCH * SEQ * E_DIM) return;
  const int e = idx & 511;
  const int s = (idx >> 9) % SEQ;
  const int b = idx / (SEQ * E_DIM);
  const int t = s / 3, tok = s - 3 * t;
  float v = X[idx];
  if (tok == 2 && t == TT - 1) v = 0.0f;
  X[idx] = v + gpe[(size_t)ts[b] * E_DIM + e] + lpe[(size_t)t * E_DIM + e];
}

// gather token positions 1::3 -> [B*T, E]
__global__ void gather_kernel(const float* __restrict__ src, float* __restrict__ dst)
{
  const int idx = blockIdx.x * 256 + threadIdx.x;
  if (idx >= BBATCH * TT * E_DIM) return;
  const int e = idx & 511;
  const int row = idx >> 9;                 // b*T + t
  const int b = row >> 8, t = row & 255;
  dst[idx] = src[((size_t)b * SEQ + 3 * t + 1) * E_DIM + e];
}

// ---------------------------------------------------------------------------
// Host orchestration
// ---------------------------------------------------------------------------
extern "C" void kernel_launch(void* const* d_in, const int* in_sizes, int n_in,
                              void* d_out, int out_size, void* d_ws, size_t ws_size,
                              hipStream_t stream) {
  (void)out_size; (void)ws_size;
  // --- resolve input pointers (handle insertion-order vs sorted-pytree order)
  const float *states, *actions, *keysIn; const int* timesteps; int pbase;
  if (in_sizes[0] == BBATCH * TT * 60) {           // states first (insertion order)
    states  = (const float*)d_in[0]; timesteps = (const int*)d_in[1];
    actions = (const float*)d_in[2]; keysIn    = (const float*)d_in[3];
    pbase = 4;
  } else {                                          // sorted: actions,keys,params...,states,timesteps
    actions = (const float*)d_in[0]; keysIn = (const float*)d_in[1];
    pbase = 2;
    states    = (const float*)d_in[n_in - 2];
    timesteps = (const int*)d_in[n_in - 1];
  }
  const float* P[46];
  for (int i = 0; i < 46; ++i) P[i] = (const float*)d_in[pbase + i];
  // params leaf order (sorted dict keys, W before b):
  enum { AE0W=0, AE0B, AE1W, AE1B,
         AP0W, AP0B, AP1W, AP1B, AP2W, AP2B,
         FC1W, FC1B, FC2W, FC2B, KW, KB_, LN1B, LN1W, LN2B, LN2W,
         OW, OB, QW, QB_, VW, VB_,
         GPE, KE0W, KE0B, KE1W, KE1B,
         LNAB, LNAW, LNSB, LNSW, LPE,
         SE0W, SE0B, SE1W, SE1B,
         SP0W, SP0B, SP1W, SP1B, SP2W, SP2B };

  // --- workspace layout (floats)
  float* Xb = (float*)d_ws;
  const size_t NX = (size_t)BBATCH * SEQ * E_DIM;        // 3,145,728
  float* Hh = Xb + NX;
  float* Qx = Hh + NX;
  float* Kx = Qx + NX;
  float* Vx = Kx + NX;
  float* Yx = Vx + NX;
  float* T1 = Yx + NX;                                   // [6144, 2048]
  float* XS = T1 + 4 * NX;
  float* XG = XS + NX;                                   // [2048, 512]
  float* HA = XG + (size_t)2048 * 512;                   // [2048, 256]
  float* HB = HA + (size_t)2048 * 256;

  auto gemm = [&](const float* A, int lda, const float* Wt, const float* bias,
                  const float* R, float* C, int ldc, int M, int N, int K, int act) {
    if ((K % 32 == 0) && (N % 64 == 0) && (M % 16 == 0)) {
      const int tiles = (M / 16) * (N / 64);
      gemm_wmma_kernel<false><<<(tiles + 7) / 8, 256, 0, stream>>>(
          A, lda, Wt, bias, R, C, ldc, M, N, K, act);
    } else {
      const int tiles = ((M + 15) / 16) * ((N + 15) / 16);
      gemm_wmma_kernel<true><<<(tiles + 7) / 8, 256, 0, stream>>>(
          A, lda, Wt, bias, R, C, ldc, M, N, K, act);
    }
  };
  auto ln = [&](const float* A, const float* w, const float* b, float* O, int rows) {
    ln_kernel<<<rows, 128, 0, stream>>>(A, E_DIM, w, b, O, E_DIM);
  };

  const int BT = BBATCH * TT;      // 2048
  const int BS = BBATCH * SEQ;     // 6144

  // --- encoders (outputs interleaved into X with row stride 3E)
  gemm(states, 60, P[SE0W], P[SE0B], nullptr, HA, 256, BT, 256, 60, 1);
  gemm(HA, 256, P[SE1W], P[SE1B], nullptr, Xb + 0,       3*E_DIM, BT, 512, 256, 0);
  gemm(keysIn, 32, P[KE0W], P[KE0B], nullptr, HA, 256, BT, 256, 32, 1);
  gemm(HA, 256, P[KE1W], P[KE1B], nullptr, Xb + E_DIM,   3*E_DIM, BT, 512, 256, 0);
  gemm(actions, 12, P[AE0W], P[AE0B], nullptr, HA, 256, BT, 256, 12, 1);
  gemm(HA, 256, P[AE1W], P[AE1B], nullptr, Xb + 2*E_DIM, 3*E_DIM, BT, 512, 256, 0);
  embed_kernel<<<(BBATCH*SEQ*E_DIM + 255)/256, 256, 0, stream>>>(Xb, P[GPE], P[LPE], timesteps);

  // --- transformer blocks
  for (int l = 0; l < NLAYER; ++l) {
    const size_t wE2 = (size_t)l * E_DIM * E_DIM, vE = (size_t)l * E_DIM;
    ln(Xb, P[LN1W] + vE, P[LN1B] + vE, Hh, BS);
    gemm(Hh, E_DIM, P[QW] + wE2, P[QB_] + vE, nullptr, Qx, E_DIM, BS, E_DIM, E_DIM, 0);
    gemm(Hh, E_DIM, P[KW] + wE2, P[KB_] + vE, nullptr, Kx, E_DIM, BS, E_DIM, E_DIM, 0);
    gemm(Hh, E_DIM, P[VW] + wE2, P[VB_] + vE, nullptr, Vx, E_DIM, BS, E_DIM, E_DIM, 0);
    attn_flash_kernel<<<dim3(SEQ/16, NHEAD, BBATCH), 32, 0, stream>>>(Qx, Kx, Vx, Yx);
    gemm(Yx, E_DIM, P[OW] + wE2, P[OB] + vE, Xb, Xb, E_DIM, BS, E_DIM, E_DIM, 0);
    ln(Xb, P[LN2W] + vE, P[LN2B] + vE, Hh, BS);
    gemm(Hh, E_DIM, P[FC1W] + (size_t)l*E_DIM*2048, P[FC1B] + (size_t)l*2048,
         nullptr, T1, 2048, BS, 2048, E_DIM, 2);
    gemm(T1, 2048, P[FC2W] + (size_t)l*2048*E_DIM, P[FC2B] + vE,
         Xb, Xb, E_DIM, BS, E_DIM, 2048, 0);
    if (l == 3)
      hipMemcpyAsync(XS, Xb, NX * sizeof(float), hipMemcpyDeviceToDevice, stream);
  }

  float* out = (float*)d_out;          // action_preds [2048,12] then state_preds [2048,60]

  // --- state head (from layer-3 output)
  gather_kernel<<<(BT*E_DIM + 255)/256, 256, 0, stream>>>(XS, XG);
  ln(XG, P[LNSW], P[LNSB], XG, BT);
  gemm(XG, E_DIM, P[SP0W], P[SP0B], nullptr, HA, 256, BT, 256, 512, 1);
  gemm(HA, 256, P[SP1W], P[SP1B], nullptr, HB, 256, BT, 256, 256, 1);
  gemm(HB, 256, P[SP2W], P[SP2B], nullptr, out + (size_t)BT*12, 60, BT, 60, 256, 0);

  // --- action head (from final output)
  gather_kernel<<<(BT*E_DIM + 255)/256, 256, 0, stream>>>(Xb, XG);
  ln(XG, P[LNAW], P[LNAB], XG, BT);
  gemm(XG, E_DIM, P[AP0W], P[AP0B], nullptr, HA, 256, BT, 256, 512, 1);
  gemm(HA, 256, P[AP1W], P[AP1B], nullptr, HB, 256, BT, 256, 256, 1);
  gemm(HB, 256, P[AP2W], P[AP2B], nullptr, out, 12, BT, 12, 256, 0);
}
